// BindingAffinityPredictor_15006615732792
// MI455X (gfx1250) — compile-verified, matrix operand order checked by reading, not run on hardware
//
#include <hip/hip_runtime.h>

typedef _Float16 f16;
typedef __attribute__((ext_vector_type(16))) _Float16 v16h;
typedef __attribute__((ext_vector_type(8)))  _Float16 v8h;
typedef __attribute__((ext_vector_type(8)))  float    v8f;

#define HID     256
#define NATOM   62
#define NBOND   6
#define NNODES  10000
#define NEDGES  160000
#define MSG_KPAD 544   // 518 padded to 17*32
#define UPD_K    512

__device__ __forceinline__ v8f vzero() {
  v8f z;
#pragma unroll
  for (int j = 0; j < 8; ++j) z[j] = 0.f;
  return z;
}

__device__ __forceinline__ v8f wmma16(v16h a, v16h b, v8f c) {
  return __builtin_amdgcn_wmma_f32_16x16x32_f16(false, a, false, b, (short)0, c,
                                                false, false);
}

// A fragment (16x32 f16, M=lane&15): lane holds K = [8g,8g+8) U [16+8g,16+8g+8)
__device__ __forceinline__ v16h load_a(const f16* base, int lda, int k0, int lane) {
  const int m = lane & 15, g = (lane >> 4) & 1;
  const f16* p = base + (size_t)m * lda + k0 + 8 * g;
  v8h lo = *(const v8h*)p;
  v8h hi = *(const v8h*)(p + 16);
  v16h a;
#pragma unroll
  for (int j = 0; j < 8; ++j) { a[j] = lo[j]; a[8 + j] = hi[j]; }
  return a;
}

// B fragment from pre-swizzled weights: frag = kt*n16+nt, 32 lanes x 16 f16 contiguous
__device__ __forceinline__ v16h load_b(const f16* w, int n16, int kt, int nt, int lane) {
  const f16* p = w + (((size_t)(kt * n16 + nt) * 32 + lane) << 4);
  v8h lo = *(const v8h*)p;
  v8h hi = *(const v8h*)(p + 8);
  v16h b;
#pragma unroll
  for (int j = 0; j < 8; ++j) { b[j] = lo[j]; b[8 + j] = hi[j]; }
  return b;
}

// fp32 [K,N] row-major -> f16 fragment-ordered (mirror of A layout on columns)
__global__ void k_swizzle(const float* __restrict__ W, f16* __restrict__ Wf,
                          int Kreal, int Kpad, int N) {
  int idx = blockIdx.x * blockDim.x + threadIdx.x;
  int n16 = N >> 4;
  int nfrag = (Kpad >> 5) * n16;
  if (idx >= nfrag * 32) return;
  int lane = idx & 31, f = idx >> 5;
  int kt = f / n16, nt = f - kt * n16;
  int n = (nt << 4) + (lane & 15);
  int g = (lane >> 4) & 1;
  f16 out[16];
#pragma unroll
  for (int j = 0; j < 8; ++j) {
    int k0 = (kt << 5) + 8 * g + j;
    int k1 = k0 + 16;
    out[j]     = (f16)((k0 < Kreal) ? W[(size_t)k0 * N + n] : 0.f);
    out[8 + j] = (f16)((k1 < Kreal) ? W[(size_t)k1 * N + n] : 0.f);
  }
  f16* p = Wf + ((size_t)idx << 4);
#pragma unroll
  for (int j = 0; j < 16; ++j) p[j] = out[j];
}

__global__ __launch_bounds__(HID) void k_embed(const float* __restrict__ af,
                                               const float* __restrict__ W,
                                               const float* __restrict__ b,
                                               f16* __restrict__ x16) {
  __shared__ float s[NATOM];
  const int node = blockIdx.x, t = threadIdx.x;
  if (t < NATOM) s[t] = af[(size_t)node * NATOM + t];
  __syncthreads();
  float acc = b[t];
  for (int k = 0; k < NATOM; ++k) acc += s[k] * W[k * HID + t];
  x16[(size_t)node * HID + t] = (f16)acc;
}

__global__ void k_zero(float* __restrict__ p, int n) {
  int i = blockIdx.x * blockDim.x + threadIdx.x;
  if (i < n) p[i] = 0.f;
}

// Fused message MLP + scatter: block = 32 edges x 256 cols, 4 waves.
// Each wave owns 4 n-tiles; each B fragment feeds two M-tiles (rows 0-15, 16-31).
__global__ __launch_bounds__(128) void k_msg(
    const f16* __restrict__ x16, const int* __restrict__ src,
    const int* __restrict__ dst, const float* __restrict__ ef,
    const f16* __restrict__ w1f, const float* __restrict__ b1,
    const f16* __restrict__ w2f, const float* __restrict__ b2,
    float* __restrict__ agg) {
  __shared__ __align__(16) f16 sA[32 * MSG_KPAD];
  __shared__ __align__(16) f16 sH[32 * HID];
  __shared__ int sSrc[32], sDst[32];
  const int tid = threadIdx.x, lane = tid & 31, wave = tid >> 5;
  const int e0 = blockIdx.x << 5;
  if (tid < 32) { sSrc[tid] = src[e0 + tid]; sDst[tid] = dst[e0 + tid]; }
  __syncthreads();
  // stage gathered A tile: 32 rows x 272 dwords (544 f16)
  unsigned* aU = (unsigned*)sA;
  for (int i = tid; i < 32 * 272; i += 128) {
    int row = i / 272, c = i - row * 272;
    unsigned v;
    if (c < 128) {
      v = ((const unsigned*)(x16 + (size_t)sSrc[row] * HID))[c];
    } else if (c < 256) {
      v = ((const unsigned*)(x16 + (size_t)sDst[row] * HID))[c - 128];
    } else if (c < 259) {
      int j = (c - 256) * 2;
      union { f16 h[2]; unsigned u; } q;
      q.h[0] = (f16)ef[(size_t)(e0 + row) * NBOND + j];
      q.h[1] = (f16)ef[(size_t)(e0 + row) * NBOND + j + 1];
      v = q.u;
    } else v = 0u;
    aU[row * 272 + c] = v;
  }
  __syncthreads();
  // GEMM1: h = relu(A @ W1 + b1); wave owns cols [64*wave, 64*wave+64)
  v8f acc[2][4];
#pragma unroll
  for (int m = 0; m < 2; ++m)
#pragma unroll
    for (int t = 0; t < 4; ++t) acc[m][t] = vzero();
  for (int kt = 0; kt < MSG_KPAD / 32; ++kt) {
    v16h a0 = load_a(sA, MSG_KPAD, kt << 5, lane);
    v16h a1 = load_a(sA + 16 * MSG_KPAD, MSG_KPAD, kt << 5, lane);
#pragma unroll
    for (int t = 0; t < 4; ++t) {
      v16h b = load_b(w1f, HID / 16, kt, wave * 4 + t, lane);
      acc[0][t] = wmma16(a0, b, acc[0][t]);
      acc[1][t] = wmma16(a1, b, acc[1][t]);
    }
  }
#pragma unroll
  for (int m = 0; m < 2; ++m)
#pragma unroll
    for (int t = 0; t < 4; ++t) {
      int col = ((wave * 4 + t) << 4) + (lane & 15);
      float bias = b1[col];
#pragma unroll
      for (int r = 0; r < 8; ++r) {
        int row = (m << 4) + r + ((lane >> 4) << 3);
        float v = acc[m][t][r] + bias;
        sH[row * HID + col] = (f16)(v > 0.f ? v : 0.f);
      }
    }
  __syncthreads();
  // GEMM2: msgs = h @ W2 + b2, then atomic scatter into agg[dst]
  v8f acc2[2][4];
#pragma unroll
  for (int m = 0; m < 2; ++m)
#pragma unroll
    for (int t = 0; t < 4; ++t) acc2[m][t] = vzero();
  for (int kt = 0; kt < HID / 32; ++kt) {
    v16h a0 = load_a(sH, HID, kt << 5, lane);
    v16h a1 = load_a(sH + 16 * HID, HID, kt << 5, lane);
#pragma unroll
    for (int t = 0; t < 4; ++t) {
      v16h b = load_b(w2f, HID / 16, kt, wave * 4 + t, lane);
      acc2[0][t] = wmma16(a0, b, acc2[0][t]);
      acc2[1][t] = wmma16(a1, b, acc2[1][t]);
    }
  }
#pragma unroll
  for (int m = 0; m < 2; ++m)
#pragma unroll
    for (int t = 0; t < 4; ++t) {
      int col = ((wave * 4 + t) << 4) + (lane & 15);
      float bias = b2[col];
#pragma unroll
      for (int r = 0; r < 8; ++r) {
        int row = (m << 4) + r + ((lane >> 4) << 3);
        unsafeAtomicAdd(&agg[(size_t)sDst[row] * HID + col],
                        acc2[m][t][r] + bias);
      }
    }
}

// Fused update MLP: x' = mlp2(concat(x, agg)); block = 16 nodes
__global__ __launch_bounds__(128) void k_upd(
    const f16* __restrict__ xin, const float* __restrict__ agg,
    const f16* __restrict__ w1f, const float* __restrict__ b1,
    const f16* __restrict__ w2f, const float* __restrict__ b2,
    f16* __restrict__ xout) {
  __shared__ __align__(16) f16 sA[16 * UPD_K];
  __shared__ __align__(16) f16 sH[16 * HID];
  const int tid = threadIdx.x, lane = tid & 31, wave = tid >> 5;
  const int n0 = blockIdx.x << 4;
  unsigned* aU = (unsigned*)sA;
  for (int i = tid; i < 16 * 256; i += 128) {
    int row = i >> 8, c = i & 255;
    unsigned v;
    if (c < 128) {
      v = ((const unsigned*)(xin + (size_t)(n0 + row) * HID))[c];
    } else {
      const float* p = agg + (size_t)(n0 + row) * HID + ((c - 128) << 1);
      union { f16 h[2]; unsigned u; } q;
      q.h[0] = (f16)p[0]; q.h[1] = (f16)p[1];
      v = q.u;
    }
    aU[(row << 8) + c] = v;
  }
  __syncthreads();
  v8f acc[4];
#pragma unroll
  for (int t = 0; t < 4; ++t) acc[t] = vzero();
  for (int kt = 0; kt < UPD_K / 32; ++kt) {
    v16h a = load_a(sA, UPD_K, kt << 5, lane);
#pragma unroll
    for (int t = 0; t < 4; ++t)
      acc[t] = wmma16(a, load_b(w1f, HID / 16, kt, wave * 4 + t, lane), acc[t]);
  }
#pragma unroll
  for (int t = 0; t < 4; ++t) {
    int col = ((wave * 4 + t) << 4) + (lane & 15);
    float bias = b1[col];
#pragma unroll
    for (int r = 0; r < 8; ++r) {
      int row = r + ((lane >> 4) << 3);
      float v = acc[t][r] + bias;
      sH[row * HID + col] = (f16)(v > 0.f ? v : 0.f);
    }
  }
  __syncthreads();
  v8f acc2[4];
#pragma unroll
  for (int t = 0; t < 4; ++t) acc2[t] = vzero();
  for (int kt = 0; kt < HID / 32; ++kt) {
    v16h a = load_a(sH, HID, kt << 5, lane);
#pragma unroll
    for (int t = 0; t < 4; ++t)
      acc2[t] = wmma16(a, load_b(w2f, HID / 16, kt, wave * 4 + t, lane), acc2[t]);
  }
#pragma unroll
  for (int t = 0; t < 4; ++t) {
    int col = ((wave * 4 + t) << 4) + (lane & 15);
    float bias = b2[col];
#pragma unroll
    for (int r = 0; r < 8; ++r) {
      int row = r + ((lane >> 4) << 3);
      xout[(size_t)(n0 + row) * HID + col] = (f16)(acc2[t][r] + bias);
    }
  }
}

// Generic relu(A@W+b) FFN, M=16 rows per block, A read straight from global
__global__ __launch_bounds__(128) void k_ffn(
    const f16* __restrict__ A, const f16* __restrict__ wf,
    const float* __restrict__ bias, f16* __restrict__ out, int K, int N) {
  const int tid = threadIdx.x, lane = tid & 31, wave = tid >> 5;
  const int m0 = blockIdx.x << 4;
  const int n16 = N >> 4;
  const int per = n16 >> 2;
  const f16* Arow = A + (size_t)m0 * K;
  for (int t = 0; t < per; ++t) {
    v8f acc = vzero();
    int nt = wave * per + t;
    for (int kt = 0; kt < (K >> 5); ++kt) {
      v16h a = load_a(Arow, K, kt << 5, lane);
      acc = wmma16(a, load_b(wf, n16, kt, nt, lane), acc);
    }
    int col = (nt << 4) + (lane & 15);
    float bb = bias[col];
#pragma unroll
    for (int r = 0; r < 8; ++r) {
      int row = r + ((lane >> 4) << 3);
      float v = acc[r] + bb;
      out[(size_t)(m0 + row) * N + col] = (f16)(v > 0.f ? v : 0.f);
    }
  }
}

__global__ void k_out_init(float* out, const float* b3) { out[0] = b3[0]; }

__global__ __launch_bounds__(256) void k_reduce(const f16* __restrict__ h2,
                                                const float* __restrict__ w3,
                                                float* __restrict__ out) {
  __shared__ float sw[128];
  __shared__ float red[256];
  const int tid = threadIdx.x;
  if (tid < 128) sw[tid] = w3[tid];
  __syncthreads();
  int node = blockIdx.x * 256 + tid;
  float v = 0.f;
  if (node < NNODES) {
    const f16* row = h2 + (size_t)node * 128;
    for (int k = 0; k < 128; ++k) v += (float)row[k] * sw[k];
  }
  red[tid] = v;
  __syncthreads();
  for (int s = 128; s > 0; s >>= 1) {
    if (tid < s) red[tid] += red[tid + s];
    __syncthreads();
  }
  if (tid == 0) unsafeAtomicAdd(out, red[0] * (1.f / NNODES));
}

extern "C" void kernel_launch(void* const* d_in, const int* in_sizes, int n_in,
                              void* d_out, int out_size, void* d_ws, size_t ws_size,
                              hipStream_t stream) {
  (void)in_sizes; (void)n_in; (void)out_size; (void)ws_size;
  const float* atom = (const float*)d_in[0];
  const int*   eidx = (const int*)d_in[1];
  const float* ef   = (const float*)d_in[2];
  const float* embW = (const float*)d_in[3];
  const float* embB = (const float*)d_in[4];
  const float* mw1  = (const float*)d_in[5];
  const float* mb1  = (const float*)d_in[6];
  const float* mw2  = (const float*)d_in[7];
  const float* mb2  = (const float*)d_in[8];
  const float* uw1  = (const float*)d_in[9];
  const float* ub1  = (const float*)d_in[10];
  const float* uw2  = (const float*)d_in[11];
  const float* ub2  = (const float*)d_in[12];
  const float* rw1  = (const float*)d_in[13];
  const float* rb1  = (const float*)d_in[14];
  const float* rw2  = (const float*)d_in[15];
  const float* rb2  = (const float*)d_in[16];
  const float* rw3  = (const float*)d_in[17];
  const float* rb3  = (const float*)d_in[18];
  const int* src = eidx;
  const int* dst = eidx + NEDGES;

  char* ws = (char*)d_ws;
  size_t cur = 0;
  auto alloc = [&](size_t bytes) -> void* {
    void* p = ws + cur;
    cur += (bytes + 255) & ~(size_t)255;
    return p;
  };
  f16*   xA   = (f16*)alloc((size_t)NNODES * HID * 2);
  f16*   xB   = (f16*)alloc((size_t)NNODES * HID * 2);
  float* agg  = (float*)alloc((size_t)NNODES * HID * 4);
  f16*   h1   = (f16*)alloc((size_t)NNODES * HID * 2);
  f16*   h2   = (f16*)alloc((size_t)NNODES * 128 * 2);
  f16*   mw1f = (f16*)alloc((size_t)6 * 272 * 512 * 2);  // 17kt*16nt frags
  f16*   mw2f = (f16*)alloc((size_t)6 * 128 * 512 * 2);  // 8kt*16nt
  f16*   uw1f = (f16*)alloc((size_t)6 * 256 * 512 * 2);  // 16kt*16nt
  f16*   uw2f = (f16*)alloc((size_t)6 * 128 * 512 * 2);
  f16*   rw1f = (f16*)alloc((size_t)128 * 512 * 2);
  f16*   rw2f = (f16*)alloc((size_t)64 * 512 * 2);       // 8kt*8nt

  auto swz = [&](const float* W, f16* Wf, int Kreal, int Kpad, int N) {
    int threads = (Kpad / 32) * (N / 16) * 32;
    k_swizzle<<<(threads + 255) / 256, 256, 0, stream>>>(W, Wf, Kreal, Kpad, N);
  };
  for (int l = 0; l < 6; ++l) {
    swz(mw1 + (size_t)l * 518 * HID, mw1f + (size_t)l * 272 * 512, 518, MSG_KPAD, HID);
    swz(mw2 + (size_t)l * HID * HID, mw2f + (size_t)l * 128 * 512, 256, 256, HID);
    swz(uw1 + (size_t)l * 512 * HID, uw1f + (size_t)l * 256 * 512, 512, 512, HID);
    swz(uw2 + (size_t)l * HID * HID, uw2f + (size_t)l * 128 * 512, 256, 256, HID);
  }
  swz(rw1, rw1f, 256, 256, 256);
  swz(rw2, rw2f, 256, 256, 128);

  k_embed<<<NNODES, HID, 0, stream>>>(atom, embW, embB, xA);

  f16* xc = xA; f16* xn = xB;
  for (int l = 0; l < 6; ++l) {
    k_zero<<<(NNODES * HID + 255) / 256, 256, 0, stream>>>(agg, NNODES * HID);
    k_msg<<<NEDGES / 32, 128, 0, stream>>>(
        xc, src, dst, ef,
        mw1f + (size_t)l * 272 * 512, mb1 + (size_t)l * HID,
        mw2f + (size_t)l * 128 * 512, mb2 + (size_t)l * HID, agg);
    k_upd<<<NNODES / 16, 128, 0, stream>>>(
        xc, agg,
        uw1f + (size_t)l * 256 * 512, ub1 + (size_t)l * HID,
        uw2f + (size_t)l * 128 * 512, ub2 + (size_t)l * HID, xn);
    f16* t = xc; xc = xn; xn = t;
  }
  k_ffn<<<NNODES / 16, 128, 0, stream>>>(xc, rw1f, rb1, h1, 256, 256);
  k_ffn<<<NNODES / 16, 128, 0, stream>>>(h1, rw2f, rb2, h2, 256, 128);
  k_out_init<<<1, 1, 0, stream>>>((float*)d_out, rb3);
  k_reduce<<<(NNODES + 255) / 256, 256, 0, stream>>>(h2, rw3, (float*)d_out);
}
